// MLP_geometry_66657892434027
// MI455X (gfx1250) — matrix operand order repaired and checked by reading, not compiled, not run on hardware
//
#include <hip/hip_runtime.h>
#include <stdint.h>

typedef __attribute__((ext_vector_type(16))) _Float16 v16h;
typedef __attribute__((ext_vector_type(8)))  float    v8f;
typedef __attribute__((ext_vector_type(2)))  __fp16   fp16x2;

#define N_LAYERS 5
// actual (unpadded) input dims of pts_W layers (layer 3 = concat, K-permuted to [h,emb])
__constant__ const int kInAct[N_LAYERS] = {93, 128, 128, 221, 128};
// K-chunks of 32 after padding: 96,128,128,224,128
// packed pts weight base offsets (dwords): tilechunks*256 cumulative (24,32,32,56,32 tilechunks)
#define PTS_B0 0
#define PTS_B1 6144
#define PTS_B2 14336
#define PTS_B3 22528
#define PTS_B4 36864
#define PTS_TOTAL_DW 45056
#define FC_BASE_DW   45056        // + l*2048
#define PACK_TOTAL_DW 55296       // 45056 + 5*8*256

// LDS column pads (dwords per point-column); pairs = rows/2
#define PC  116    // bufC: 112 pairs used (layer3: h pairs 0..63, emb pairs 64..111)
#define PH  68     // bufH: 64 pairs used
#define PCB 20     // cbuf: 16 pairs used
#define WAVE_DW (16 * PC + 16 * PH + 16 * PCB)   // 3264 dwords = 13056 B per wave

__device__ __forceinline__ unsigned pack2(float a, float b) {
    union { fp16x2 h; unsigned u; } c;
    c.h = __builtin_amdgcn_cvt_pkrtz(a, b);
    return c.u;
}

struct PackParams {
    const float* w[10];           // pts_W[0..4], fc_W[0..4]
    unsigned int* packed;
};

// Pack fp32 weights into the CDNA5 16-bit A-matrix (16x32) per-lane layout:
// lane L holds M=L%16 ; vgpr v holds K = kc*32 + (v>=4?16:0) + (L>=16?8:0) + (v%4)*2 .. +1
// Layer 3 (concat) K is permuted: packed kk<128 -> source col 93+kk (h part),
// 128<=kk<221 -> source col kk-128 (emb part), else zero.
__global__ __launch_bounds__(256) void pack_weights_kernel(PackParams pp) {
    int tid = blockIdx.x * 256 + threadIdx.x;
    if (tid >= PACK_TOTAL_DW) return;

    const float* W;
    int IN, kc, m, lane, v;
    bool isL3 = false;
    if (tid < PTS_TOTAL_DW) {
        const int base[6] = {PTS_B0, PTS_B1, PTS_B2, PTS_B3, PTS_B4, PTS_TOTAL_DW};
        int l = 0;
        while (tid >= base[l + 1]) ++l;
        int rem = tid - base[l];
        int tc  = rem >> 8;
        lane    = (rem >> 3) & 31;
        v       = rem & 7;
        kc      = tc >> 3;
        m       = tc & 7;
        W       = pp.w[l];
        IN      = kInAct[l];
        isL3    = (l == 3);
    } else {
        int t2  = tid - FC_BASE_DW;
        int l   = t2 >> 11;           // 2048 dwords per layer
        int rem = t2 & 2047;
        int tc  = rem >> 8;
        lane    = (rem >> 3) & 31;
        v       = rem & 7;
        kc      = 0;
        m       = tc;
        W       = pp.w[5 + l];
        IN      = 32;
    }
    int M = m * 16 + (lane & 15);
    int K = kc * 32 + ((v & 4) ? 16 : 0) + ((lane & 16) ? 8 : 0) + ((v & 3) << 1);

    int s0, s1;
    if (isL3) {
        s0 = (K     < 128) ? 93 + K     : ((K     < 221) ? K     - 128 : -1);
        s1 = (K + 1 < 128) ? 93 + K + 1 : ((K + 1 < 221) ? K + 1 - 128 : -1);
    } else {
        s0 = (K     < IN) ? K     : -1;
        s1 = (K + 1 < IN) ? K + 1 : -1;
    }
    float f0 = (s0 >= 0) ? W[M * IN + s0] : 0.0f;
    float f1 = (s1 >= 0) ? W[M * IN + s1] : 0.0f;
    pp.packed[tid] = pack2(f0, f1);
}

struct Params {
    const float* p;
    const float* cloud_pos;
    const float* feats;
    const float* c_fallback;
    const float* B;                // (3,93)
    const float* pts_b[N_LAYERS];
    const float* fc_b[N_LAYERS];
    const float* out_W;            // (1,128)
    const float* out_b;            // (1)
    const int*   I;                // (N,8)
    const int*   nbr_num;          // (N)
    const unsigned int* packed;
    float* out;                    // (N)
    float* mask;                   // (N/16)
    float* has_nb;                 // (N)
};

// B-matrix chunk (32 K-rows x 16 points) from pair-packed column-major LDS:
// lane needs pairs pairOff + kc*16 + hf*4 + v (v=0..3) and pairOff + kc*16 + 8 + hf*4 + v of its column -> 2x ds_load_b128.
__device__ __forceinline__ v16h load_b(const unsigned int* col, int pairOff, int kc, int hf) {
    const unsigned int* ptr = col + pairOff + kc * 16 + hf * 4;
    union { uint4 q[2]; v16h v; } u;
    u.q[0] = *(const uint4*)ptr;
    u.q[1] = *(const uint4*)(ptr + 8);
    return u.v;
}

__device__ __forceinline__ v16h load_a(const unsigned int* packed, int baseDw, int tc, int lane) {
    const unsigned int* ptr = packed + baseDw + tc * 256 + lane * 8;
    return *(const v16h*)ptr;   // 32B contiguous per lane -> global_load_b128 x2
}

__global__ __launch_bounds__(128) void mlp_geometry_kernel(Params prm) {
    const int lane = threadIdx.x & 31;
    const int wv   = threadIdx.x >> 5;
    const int g    = blockIdx.x * 4 + wv;          // group of 16 points per wave
    const int N    = lane & 15;
    const int hf   = (lane >> 4) & 1;
    const int n    = g * 16 + N;                   // point index (lanes L and L+16 share n)

    __shared__ unsigned int smem[4 * WAVE_DW];
    unsigned int* bufC = smem + wv * WAVE_DW;
    unsigned int* bufH = bufC + 16 * PC;
    unsigned int* cbuf = bufH + 16 * PH;
    unsigned int* colC = bufC + N * PC;            // this lane's point-column
    unsigned int* colH = bufH + N * PH;
    unsigned int* colB = cbuf + N * PCB;

    // ---------- phase 1: neighbor gather + distance-weighted feature blend ----------
    const float px = prm.p[n * 3 + 0];
    const float py = prm.p[n * 3 + 1];
    const float pz = prm.p[n * 3 + 2];

    float wk[8];
    int   idx[8];
    float wsum = 0.0f;
#pragma unroll
    for (int k = 0; k < 8; ++k) {
        int id = prm.I[n * 8 + k];
        idx[k] = id;
        float dx = prm.cloud_pos[id * 3 + 0] - px;
        float dy = prm.cloud_pos[id * 3 + 1] - py;
        float dz = prm.cloud_pos[id * 3 + 2] - pz;
        float D  = dx * dx + dy * dy + dz * dz;
        bool  far = D > 1.0f;
        D = far ? 1.0e4f : D;
        float w = far ? 0.0f : 1.0f / (D + 1.0e-10f);
        wsum += w;
        wk[k] = w;
    }
    const float inv = 1.0f / fmaxf(wsum, 1.0e-12f);
    const bool  has = prm.nbr_num[n] > 1;

    const int fb = hf * 16;               // this lane handles features fb..fb+15
    float cv[16];
#pragma unroll
    for (int j = 0; j < 16; ++j) cv[j] = 0.0f;
#pragma unroll
    for (int k = 0; k < 8; ++k) {
        float wn = wk[k] * inv;
        const float* f = prm.feats + idx[k] * 32 + fb;
#pragma unroll
        for (int j = 0; j < 16; ++j) cv[j] += wn * f[j];
    }
    if (!has) {
#pragma unroll
        for (int j = 0; j < 16; ++j) cv[j] = prm.c_fallback[fb + j];
    }
    {   // pair-pack 16 features -> 8 dwords -> 2x ds_store_b128 at pairs hf*8..hf*8+7
        unsigned d[8];
#pragma unroll
        for (int j = 0; j < 8; ++j) d[j] = pack2(cv[2 * j], cv[2 * j + 1]);
        *(uint4*)(colB + hf * 8)     = make_uint4(d[0], d[1], d[2], d[3]);
        *(uint4*)(colB + hf * 8 + 4) = make_uint4(d[4], d[5], d[6], d[7]);
    }

    // has_neighbors + valid_ray_mask (pts_num == 16 == one wave's group)
    unsigned long long bal = __ballot(has ? 1 : 0);
    if (lane == 0) {
        int cnt = __popcll(bal & 0xFFFFull);
        prm.mask[g] = (cnt >= 9) ? 1.0f : 0.0f;
    }
    if (lane < 16) prm.has_nb[n] = has ? 1.0f : 0.0f;

    // ---------- phase 2: fourier embedding at bufC pairs 64..111 (rows 93..95 zero) ----------
    {
        const int e0 = hf * 48;   // hf0: emb rows 0..47 ; hf1: rows 48..95 (93..95 -> 0)
#pragma unroll
        for (int t = 0; t < 6; ++t) {
            unsigned dd[4];
#pragma unroll
            for (int q = 0; q < 4; ++q) {
                int e = e0 + t * 8 + 2 * q;
                float s0 = 0.0f, s1 = 0.0f;
                if (e < 93)
                    s0 = __sinf(6.283185307179586f *
                                (px * prm.B[e] + py * prm.B[93 + e] + pz * prm.B[186 + e]));
                if (e + 1 < 93)
                    s1 = __sinf(6.283185307179586f *
                                (px * prm.B[e + 1] + py * prm.B[94 + e] + pz * prm.B[187 + e]));
                dd[q] = pack2(s0, s1);
            }
            *(uint4*)(colC + 64 + hf * 24 + t * 4) = make_uint4(dd[0], dd[1], dd[2], dd[3]);
        }
    }
    __syncthreads();

    // fc B operand (c features, K=32) is invariant across layers: load once
    const v16h cB = load_b(colB, 0, 0, hf);

    const int chunksArr[N_LAYERS] = {3, 4, 4, 7, 4};
    const int ptsBase[N_LAYERS]   = {PTS_B0, PTS_B1, PTS_B2, PTS_B3, PTS_B4};
    const int pairOff[N_LAYERS]   = {64, 0, 0, 0, 0};   // layer0 reads emb region of bufC

    // ---------- phase 3: 5-layer skip-MLP via WMMA ----------
    for (int l = 0; l < N_LAYERS; ++l) {
        const unsigned int* colIn = (l == 0 || l == 3) ? colC : colH;
        const int chunks = chunksArr[l];
        const int pbase  = ptsBase[l];
        const int fbase  = FC_BASE_DW + l * 2048;
        const int poff   = pairOff[l];
        const float* pb  = prm.pts_b[l];
        const float* fcb = prm.fc_b[l];

        v8f acc[8];
        const v8f zf = {0.f, 0.f, 0.f, 0.f, 0.f, 0.f, 0.f, 0.f};
#pragma unroll
        for (int m = 0; m < 8; ++m) acc[m] = zf;

        for (int kc = 0; kc < chunks; ++kc) {
            v16h b = load_b(colIn, poff, kc, hf);
#pragma unroll
            for (int m = 0; m < 8; ++m) {
                v16h a = load_a(prm.packed, pbase, kc * 8 + m, lane);
                acc[m] = __builtin_amdgcn_wmma_f32_16x16x32_f16(
                    false, a, false, b, (short)0, acc[m], false, false);
            }
        }

#pragma unroll
        for (int m = 0; m < 8; ++m) {
            const int row0 = m * 16 + hf * 8;
#pragma unroll
            for (int r = 0; r < 8; ++r)
                acc[m][r] = fmaxf(acc[m][r] + pb[row0 + r], 0.0f);   // bias + ReLU
            v16h fa = load_a(prm.packed, fbase, m, lane);
            acc[m] = __builtin_amdgcn_wmma_f32_16x16x32_f16(
                false, fa, false, cB, (short)0, acc[m], false, false); // h += c @ fcW.T
#pragma unroll
            for (int r = 0; r < 8; ++r)
                acc[m][r] += fcb[row0 + r];
        }

        // store layer output: pair-pack -> 1x ds_store_b128 per tile.
        // Skip layer (l==2) writes h into bufC pairs 0..63 ([h,emb] permuted concat).
        unsigned int* dcol = (l == 2) ? colC : colH;
#pragma unroll
        for (int m = 0; m < 8; ++m) {
            unsigned d0 = pack2(acc[m][0], acc[m][1]);
            unsigned d1 = pack2(acc[m][2], acc[m][3]);
            unsigned d2 = pack2(acc[m][4], acc[m][5]);
            unsigned d3 = pack2(acc[m][6], acc[m][7]);
            *(uint4*)(dcol + m * 8 + hf * 4) = make_uint4(d0, d1, d2, d3);
        }
        __syncthreads();
    }

    // ---------- phase 4: out = h @ out_W.T + out_b (1x128 dot, split across lane pairs) ----------
    float sum = 0.0f;
    const int kb = hf * 64;
#pragma unroll
    for (int t = 0; t < 8; ++t) {
        union { uint4 q; _Float16 h[8]; } u;
        u.q = *(const uint4*)(colH + hf * 32 + t * 4);
#pragma unroll
        for (int i = 0; i < 8; ++i)
            sum += prm.out_W[kb + 8 * t + i] * (float)u.h[i];
    }
    sum += __shfl_xor(sum, 16);
    if (lane < 16) prm.out[n] = sum + prm.out_b[0];
}

extern "C" void kernel_launch(void* const* d_in, const int* in_sizes, int n_in,
                              void* d_out, int out_size, void* d_ws, size_t ws_size,
                              hipStream_t stream) {
    const float* ptsW[N_LAYERS];
    const float* ptsb[N_LAYERS];
    const float* fcW[N_LAYERS];
    const float* fcb[N_LAYERS];
    const float *p, *cloud, *feats, *cfb, *Bp, *outW, *outb;
    const int *I, *nn;

    if (n_in >= 30) {
        p = (const float*)d_in[0]; cloud = (const float*)d_in[1];
        feats = (const float*)d_in[2]; cfb = (const float*)d_in[3];
        Bp = (const float*)d_in[4];
        for (int i = 0; i < 5; ++i) ptsW[i] = (const float*)d_in[5 + i];
        for (int i = 0; i < 5; ++i) ptsb[i] = (const float*)d_in[10 + i];
        for (int i = 0; i < 5; ++i) fcW[i]  = (const float*)d_in[15 + i];
        for (int i = 0; i < 5; ++i) fcb[i]  = (const float*)d_in[20 + i];
        outW = (const float*)d_in[25]; outb = (const float*)d_in[26];
        I = (const int*)d_in[27]; nn = (const int*)d_in[28];
    } else {
        p = (const float*)d_in[0]; cloud = (const float*)d_in[1];
        feats = (const float*)d_in[2]; cfb = (const float*)d_in[3];
        Bp = (const float*)d_in[4];
        const float* pw  = (const float*)d_in[5];
        const float* pbv = (const float*)d_in[6];
        const float* fw  = (const float*)d_in[7];
        const float* fbv = (const float*)d_in[8];
        const int inAct[5] = {93, 128, 128, 221, 128};
        size_t off = 0;
        for (int i = 0; i < 5; ++i) { ptsW[i] = pw + off; off += (size_t)128 * inAct[i]; }
        for (int i = 0; i < 5; ++i) ptsb[i] = pbv + (size_t)i * 128;
        for (int i = 0; i < 5; ++i) fcW[i]  = fw  + (size_t)i * 128 * 32;
        for (int i = 0; i < 5; ++i) fcb[i]  = fbv + (size_t)i * 128;
        outW = (const float*)d_in[9]; outb = (const float*)d_in[10];
        I = (const int*)d_in[11]; nn = (const int*)d_in[12];
    }

    const int N = in_sizes[0] / 3;            // 262144
    unsigned int* packed = (unsigned int*)d_ws;

    // 1) pack weights to f16 WMMA A-matrix layout in workspace (layer-3 K permuted)
    PackParams pp;
    for (int i = 0; i < 5; ++i) pp.w[i] = ptsW[i];
    for (int i = 0; i < 5; ++i) pp.w[5 + i] = fcW[i];
    pp.packed = packed;
    pack_weights_kernel<<<(PACK_TOTAL_DW + 255) / 256, 256, 0, stream>>>(pp);

    // 2) fused gather + fourier + WMMA MLP
    Params prm;
    prm.p = p; prm.cloud_pos = cloud; prm.feats = feats; prm.c_fallback = cfb;
    prm.B = Bp;
    for (int i = 0; i < 5; ++i) { prm.pts_b[i] = ptsb[i]; prm.fc_b[i] = fcb[i]; }
    prm.out_W = outW; prm.out_b = outb;
    prm.I = I; prm.nbr_num = nn;
    prm.packed = packed;
    float* out = (float*)d_out;
    prm.out = out;
    prm.mask = out + N;
    prm.has_nb = out + N + N / 16;

    const int groups = N / 16;                // one wave per 16 points, 4 waves/block
    mlp_geometry_kernel<<<groups / 4, 128, 0, stream>>>(prm);
}
